// Encoder_65721589563981
// MI455X (gfx1250) — compile-verified, matrix operand order checked by reading, not run on hardware
//
#include <hip/hip_runtime.h>

#define TT 256
#define BB 64
#define DD 512
#define UU 512
#define G3 1536
#define HROW 520   // 512 + 8 halves pad -> 1040B row stride, bank-conflict-free b128 reads

typedef __attribute__((ext_vector_type(16))) _Float16 v16h;
typedef __attribute__((ext_vector_type(8)))  _Float16 v8h;
typedef __attribute__((ext_vector_type(4)))  _Float16 v4h;
typedef __attribute__((ext_vector_type(8)))  float    v8f;

// ---- WMMA fragment loader (16-bit A layout; also B via pre-transposed weights) ----
// A 16x32 f16: lane(0-15)=row M, VGPR0-3 = K0..7, VGPR4-7 = K16..23; lanes 16-31 same
// rows with K8..15 / K24..31. B 32x16 from col-major storage has the mirrored layout,
// so one loader serves both: two 16B loads per lane.
__device__ __forceinline__ v16h load_frag(const _Float16* tile, int row_stride, int lane) {
  const _Float16* p = tile + (size_t)(lane & 15) * row_stride + ((lane >> 4) & 1) * 8;
  v8h c0 = *(const v8h*)(p);
  v8h c1 = *(const v8h*)(p + 16);
  v16h r;
#pragma unroll
  for (int i = 0; i < 8; ++i) { r[i] = c0[i]; r[8 + i] = c1[i]; }
  return r;
}

__device__ __forceinline__ float sigf(float x)  { return 1.0f / (1.0f + __expf(-x)); }
__device__ __forceinline__ float tanhf_(float x){ float e = __expf(2.0f * x); return 1.0f - 2.0f / (e + 1.0f); }

// ---- Kernel 0a: embedding gather + f32->f16, e16[row = t*64+b][512] ----
__global__ void __launch_bounds__(128) k_embed(const int* __restrict__ x,
                                               const float* __restrict__ emb,
                                               _Float16* __restrict__ e16) {
  int row = blockIdx.x;            // 0..16383, row = t*64 + b
  int b = row & 63, t = row >> 6;
  int tok = x[b * TT + t];
  int c = threadIdx.x * 4;         // 128 threads * 4 = 512
  float4 v = *(const float4*)(emb + (size_t)tok * DD + c);
  v4h h; h[0] = (_Float16)v.x; h[1] = (_Float16)v.y; h[2] = (_Float16)v.z; h[3] = (_Float16)v.w;
  *(v4h*)(e16 + (size_t)row * DD + c) = h;
}

// ---- Kernel 0b: transpose-convert 4 weight mats [512][1536]f32 -> [1536][512]f16 ----
__global__ void __launch_bounds__(128) k_cvtT(const float* __restrict__ Wf, const float* __restrict__ Wb,
                                              const float* __restrict__ Uf, const float* __restrict__ Ub,
                                              _Float16* __restrict__ dstbase) {
  int mat = blockIdx.x / G3;
  int n   = blockIdx.x % G3;
  const float* src = (mat == 0) ? Wf : (mat == 1) ? Wb : (mat == 2) ? Uf : Ub;
  _Float16* dst = dstbase + (size_t)mat * (G3 * DD) + (size_t)n * DD;
  for (int k = threadIdx.x; k < DD; k += 128)
    dst[k] = (_Float16)src[(size_t)k * G3 + n];
}

// ---- Kernel 1: gx = e @ W + b_in, per direction, stored f16 [row][1536] ----
// Each wave computes one 16-row x 32-col strip (2 column tiles reusing one A frag).
__global__ void __launch_bounds__(256) k_gx(const _Float16* __restrict__ e16,
                                            const _Float16* __restrict__ W16T, // [dir][1536][512]
                                            const float* __restrict__ bf, const float* __restrict__ bb,
                                            _Float16* __restrict__ gx /* [dir][16384][1536] */) {
  int lane = threadIdx.x & 31;
  int wave = threadIdx.x >> 5;                     // 8 waves
  int strip = blockIdx.x * 8 + wave;               // 2*1024*48 strips
  int dir = strip / (1024 * 48);
  int rem = strip % (1024 * 48);
  int rt = rem / 48, cp = rem % 48;                // row tile, column pair
  const _Float16* WT = W16T + (size_t)dir * (G3 * DD);
  const float* bin = dir ? bb : bf;                // bias[0]
  _Float16* g = gx + (size_t)dir * (16384ull * G3);
  v8f acc0 = {}, acc1 = {};
  for (int k = 0; k < 16; ++k) {
    v16h a  = load_frag(e16 + (size_t)(rt * 16) * DD + k * 32, DD, lane);
    v16h b0 = load_frag(WT  + (size_t)(cp * 32)      * DD + k * 32, DD, lane);
    v16h b1 = load_frag(WT  + (size_t)(cp * 32 + 16) * DD + k * 32, DD, lane);
    acc0 = __builtin_amdgcn_wmma_f32_16x16x32_f16(false, a, false, b0, (short)0, acc0, false, false);
    acc1 = __builtin_amdgcn_wmma_f32_16x16x32_f16(false, a, false, b1, (short)0, acc1, false, false);
  }
  int n0 = cp * 32 + (lane & 15);
  float bi0 = bin[n0], bi1 = bin[n0 + 16];
  int hi = (lane >> 4) & 1;
#pragma unroll
  for (int i = 0; i < 8; ++i) {
    size_t row = rt * 16 + hi * 8 + i;
    g[row * G3 + n0]      = (_Float16)(acc0[i] + bi0);
    g[row * G3 + n0 + 16] = (_Float16)(acc1[i] + bi1);
  }
}

// ---- Kernel 2: persistent GRU recurrence. Batch rows are independent, so split
// M=64 into 4 groups of 16 (one WMMA M-tile) per direction: 8 workgroups, no
// cross-WG communication, 4x shorter per-step critical path. ----
__global__ void __launch_bounds__(512) k_gru(const _Float16* __restrict__ U16T, // [2][1536][512]
                                             const float* __restrict__ bf, const float* __restrict__ bb,
                                             const _Float16* __restrict__ gx, // [2][16384][1536]
                                             const int* __restrict__ x,
                                             float* __restrict__ out,         // [64][256][1024]
                                             float* __restrict__ state) {     // [64][1024]
  __shared__ _Float16 h16[2][16][HROW];            // ~33 KB double-buffered state tile
  __shared__ float    msk[2][16];
  const int dir = blockIdx.x >> 2;                 // 0..1
  const int mg  = blockIdx.x & 3;                  // batch group: rows mg*16 .. mg*16+15
  const _Float16* UT  = U16T + (size_t)dir * (G3 * DD);
  const float*    brc = (dir ? bb : bf) + G3;      // bias[1] (recurrent)
  const _Float16* gxd = gx + (size_t)dir * (16384ull * G3);
  const int tid = threadIdx.x, lane = tid & 31, wave = tid >> 5; // 16 waves
  const int l15 = lane & 15, hi = (lane >> 4) & 1;

  for (int i = tid; i < 2 * 16 * HROW; i += 512) ((_Float16*)h16)[i] = (_Float16)0.0f;
  float hreg[2][8] = {};                           // private f32 hidden state (own u-slice)
  __syncthreads();

  for (int s = 0; s < TT; ++s) {
    const int t = dir ? (TT - 1 - s) : s;
    if (tid < 16) msk[s & 1][tid] = (x[(mg * 16 + tid) * TT + t] != 0) ? 1.0f : 0.0f;
    __syncthreads();
    const int rb = s & 1, wb = rb ^ 1;
    if (s + 1 < TT) { // prefetch next step's gx rows for this batch group
      int tn = dir ? (TT - 2 - s) : (s + 1);
      __builtin_prefetch(gxd + ((size_t)tn * BB + mg * 16 + (tid & 15)) * G3 + ((tid >> 4) & 7) * 192, 0, 1);
    }
#pragma unroll
    for (int jj = 0; jj < 2; ++jj) {
      const int j = wave * 2 + jj;                 // u-tile 0..31
      const int ucol = j * 16 + l15;
      const float br0 = brc[ucol], br1 = brc[UU + ucol], br2 = brc[2 * UU + ucol];
      v8f acc[3] = {};
      for (int k = 0; k < 16; ++k) {
        v16h bfr[3];
#pragma unroll
        for (int g = 0; g < 3; ++g)
          bfr[g] = load_frag(UT + (size_t)(g * UU + j * 16) * DD + k * 32, DD, lane);
        v16h a = load_frag(&h16[rb][0][k * 32], HROW, lane);
#pragma unroll
        for (int g = 0; g < 3; ++g)
          acc[g] = __builtin_amdgcn_wmma_f32_16x16x32_f16(false, a, false, bfr[g],
                                                          (short)0, acc[g], false, false);
      }
#pragma unroll
      for (int i = 0; i < 8; ++i) {
        const int bl = hi * 8 + i;                 // local row in the 16-row tile
        const int b  = mg * 16 + bl;
        const size_t ro = ((size_t)t * BB + b) * G3;
        float gxz = (float)gxd[ro + ucol];
        float gxr = (float)gxd[ro + UU + ucol];
        float gxh = (float)gxd[ro + 2 * UU + ucol];
        float z  = sigf(gxz + acc[0][i] + br0);
        float r  = sigf(gxr + acc[1][i] + br1);
        float hc = tanhf_(gxh + r * (acc[2][i] + br2));
        float hp = hreg[jj][i];
        float hn = z * hp + (1.0f - z) * hc;
        hn = hp + msk[s & 1][bl] * (hn - hp);      // masked step keeps state
        hreg[jj][i] = hn;
        out[((size_t)b * TT + t) * 1024 + dir * UU + ucol] = hn; // out[t] == h (Keras mask)
        h16[wb][bl][ucol] = (_Float16)hn;
      }
    }
    __syncthreads();
  }
#pragma unroll
  for (int jj = 0; jj < 2; ++jj) {
    const int ucol = (wave * 2 + jj) * 16 + l15;
#pragma unroll
    for (int i = 0; i < 8; ++i)
      state[(size_t)(mg * 16 + hi * 8 + i) * 1024 + dir * UU + ucol] = hreg[jj][i];
  }
}

// ---- Kernel 3: LayerNorm in place over d_out rows of 1024 ----
__global__ void __launch_bounds__(256) k_ln(float* __restrict__ out,
                                            const float* __restrict__ gamma,
                                            const float* __restrict__ beta) {
  __shared__ float red[2][8];
  float* p = out + (size_t)blockIdx.x * 1024;
  int c = threadIdx.x * 4;
  float4 v = *(const float4*)(p + c);
  float s = v.x + v.y + v.z + v.w;
  float q = v.x * v.x + v.y * v.y + v.z * v.z + v.w * v.w;
#pragma unroll
  for (int off = 16; off > 0; off >>= 1) { s += __shfl_xor(s, off); q += __shfl_xor(q, off); }
  int lane = threadIdx.x & 31, wave = threadIdx.x >> 5;
  if (lane == 0) { red[0][wave] = s; red[1][wave] = q; }
  __syncthreads();
  float ts = 0.f, tq = 0.f;
#pragma unroll
  for (int w = 0; w < 8; ++w) { ts += red[0][w]; tq += red[1][w]; }
  float mu = ts * (1.0f / 1024.0f);
  float var = tq * (1.0f / 1024.0f) - mu * mu;
  float rs = rsqrtf(var + 0.001f);
  float4 g4 = *(const float4*)(gamma + c);
  float4 b4 = *(const float4*)(beta + c);
  v.x = (v.x - mu) * rs * g4.x + b4.x;
  v.y = (v.y - mu) * rs * g4.y + b4.y;
  v.z = (v.z - mu) * rs * g4.z + b4.z;
  v.w = (v.w - mu) * rs * g4.w + b4.w;
  *(float4*)(p + c) = v;
}

extern "C" void kernel_launch(void* const* d_in, const int* in_sizes, int n_in,
                              void* d_out, int out_size, void* d_ws, size_t ws_size,
                              hipStream_t stream) {
  const int*   x     = (const int*)d_in[0];
  const float* emb   = (const float*)d_in[1];
  const float* W_fwd = (const float*)d_in[2];
  const float* U_fwd = (const float*)d_in[3];
  const float* b_fwd = (const float*)d_in[4];
  const float* W_bwd = (const float*)d_in[5];
  const float* U_bwd = (const float*)d_in[6];
  const float* b_bwd = (const float*)d_in[7];
  const float* gamma = (const float*)d_in[8];
  const float* beta  = (const float*)d_in[9];
  float* out = (float*)d_out;                       // [64][256][1024] then state [64][1024]
  float* state = out + (size_t)BB * TT * 1024;

  char* ws = (char*)d_ws;
  _Float16* e16  = (_Float16*)ws;                             // 16 MB
  _Float16* WT   = (_Float16*)(ws + (16384ull * DD * 2));     // 4 x 1.5 MB: W_f, W_b, U_f, U_b
  _Float16* gx   = (_Float16*)(ws + (16384ull * DD * 2) + 4ull * G3 * DD * 2); // 2 x 48 MB
  _Float16* U16T = WT + 2ull * G3 * DD;

  k_embed<<<dim3(16384), dim3(128), 0, stream>>>(x, emb, e16);
  k_cvtT <<<dim3(4 * G3), dim3(128), 0, stream>>>(W_fwd, W_bwd, U_fwd, U_bwd, WT);
  k_gx   <<<dim3(12288), dim3(256), 0, stream>>>(e16, WT, b_fwd, b_bwd, gx);
  k_gru  <<<dim3(8), dim3(512), 0, stream>>>(U16T, b_fwd, b_bwd, gx, x, out, state);
  k_ln   <<<dim3(16384), dim3(256), 0, stream>>>(out, gamma, beta);
}
